// MoE_64209761075861
// MI455X (gfx1250) — matrix operand order repaired and checked by reading, not compiled, not run on hardware
//
#include <hip/hip_runtime.h>
#include <hip/hip_bf16.h>

// Problem constants (from reference)
#define BT    2048          // B*T tokens
#define DDIM  768
#define HDIM  1536
#define NEXP  8

#define W1N   ((size_t)NEXP * 2 * HDIM * DDIM)   // 18,874,368 elems
#define W2N   ((size_t)NEXP * DDIM * HDIM)       //  9,437,184 elems

typedef __attribute__((ext_vector_type(16))) __bf16 v16bf;
typedef __attribute__((ext_vector_type(8)))  float  v8f;

union FragU {
    v16bf v;
    uint4 q[2];
    unsigned short s[16];
};

// float -> bf16, round to nearest even
static __device__ __forceinline__ unsigned short f2bf(float f) {
    unsigned int u = __float_as_uint(f);
    unsigned int r = u + 0x7FFFu + ((u >> 16) & 1u);
    return (unsigned short)(r >> 16);
}

// A-operand fragment (16x32 bf16) from an LDS row. Contiguous k order, same as B:
// lanes 0-15 hold k {kb..kb+15}; lanes 16-31 {kb+16..kb+31}.
static __device__ __forceinline__ v16bf load_a_lds(const unsigned short* base, int k0) {
    FragU f;
    f.q[0] = *reinterpret_cast<const uint4*>(base + k0);
    f.q[1] = *reinterpret_cast<const uint4*>(base + k0 + 8);
    return f.v;
}

// B-operand fragment (32x16): this lane's column = one weight row,
// 16 consecutive K values. Pre-converted bf16: two raw 16-byte loads.
static __device__ __forceinline__ v16bf load_b_bf16(const unsigned short* p) {
    FragU f;
    f.q[0] = *reinterpret_cast<const uint4*>(p);
    f.q[1] = *reinterpret_cast<const uint4*>(p + 8);
    return f.v;
}

// Fallback: B fragment from fp32 weights, converted in-register.
static __device__ __forceinline__ v16bf load_b_f32(const float* p) {
    FragU f;
#pragma unroll
    for (int i = 0; i < 16; i += 4) {
        float4 t = *reinterpret_cast<const float4*>(p + i);
        f.s[i + 0] = f2bf(t.x);
        f.s[i + 1] = f2bf(t.y);
        f.s[i + 2] = f2bf(t.z);
        f.s[i + 3] = f2bf(t.w);
    }
    return f.v;
}

// ---------------------------------------------------------------------------
// Kernel 0: zero output + workspace counters
// ---------------------------------------------------------------------------
__global__ void moe_zero_kernel(float* out, int n, int* cnt, float* probsum) {
    int i = blockIdx.x * blockDim.x + threadIdx.x;
    int stride = gridDim.x * blockDim.x;
    for (; i < n; i += stride) out[i] = 0.0f;
    if (blockIdx.x == 0 && threadIdx.x < NEXP) {
        cnt[threadIdx.x] = 0;
        probsum[threadIdx.x] = 0.0f;
    }
}

// ---------------------------------------------------------------------------
// Kernel 0b: fp32 -> bf16 weight conversion (one pass, bandwidth bound)
// ---------------------------------------------------------------------------
__global__ void moe_conv_kernel(const float* __restrict__ src,
                                unsigned short* __restrict__ dst, int n4) {
    int i = blockIdx.x * blockDim.x + threadIdx.x;
    int stride = gridDim.x * blockDim.x;
    for (; i < n4; i += stride) {
        float4 t = reinterpret_cast<const float4*>(src)[i];
        unsigned int lo = (unsigned int)f2bf(t.x) | ((unsigned int)f2bf(t.y) << 16);
        unsigned int hi = (unsigned int)f2bf(t.z) | ((unsigned int)f2bf(t.w) << 16);
        reinterpret_cast<uint2*>(dst)[i] = make_uint2(lo, hi);
    }
}

// ---------------------------------------------------------------------------
// Kernel 1: gating. One wave (32 lanes) per token.
// ---------------------------------------------------------------------------
__global__ void moe_gate_kernel(const float* __restrict__ x,
                                const float* __restrict__ Wg,
                                const float* __restrict__ eb,
                                int* cnt, float* probsum,
                                int* list, float* gwl) {
    int wave = threadIdx.x >> 5;
    int lane = threadIdx.x & 31;
    int t = blockIdx.x * 8 + wave;
    if (t >= BT) return;

    const float* xr = x + (size_t)t * DDIM;
    float p[NEXP];
#pragma unroll
    for (int e = 0; e < NEXP; ++e) p[e] = 0.0f;

    for (int d = lane; d < DDIM; d += 32) {
        float xv = xr[d];
#pragma unroll
        for (int e = 0; e < NEXP; ++e) p[e] += xv * Wg[e * DDIM + d];
    }
#pragma unroll
    for (int e = 0; e < NEXP; ++e) {
#pragma unroll
        for (int off = 16; off > 0; off >>= 1) p[e] += __shfl_xor(p[e], off, 32);
    }

    if (lane == 0) {
        float sc[NEXP];
#pragma unroll
        for (int e = 0; e < NEXP; ++e) sc[e] = p[e] + eb[e];

        // top-2, ties -> lowest index (matches jax.lax.top_k)
        int i0 = 0;
#pragma unroll
        for (int e = 1; e < NEXP; ++e) if (sc[e] > sc[i0]) i0 = e;
        int i1 = (i0 == 0) ? 1 : 0;
#pragma unroll
        for (int e = 0; e < NEXP; ++e) if (e != i0 && e != i1 && sc[e] > sc[i1]) i1 = e;

        float e1 = __expf(sc[i1] - sc[i0]);
        float s  = 1.0f + e1;
        float w0 = 1.0f / s;
        float w1 = e1 / s;

        float M = sc[0];
#pragma unroll
        for (int e = 1; e < NEXP; ++e) M = fmaxf(M, sc[e]);
        float pe[NEXP];
        float ps = 0.0f;
#pragma unroll
        for (int e = 0; e < NEXP; ++e) { pe[e] = __expf(sc[e] - M); ps += pe[e]; }
        float inv = 1.0f / ps;
#pragma unroll
        for (int e = 0; e < NEXP; ++e) atomicAdd(&probsum[e], pe[e] * inv);

        int p0 = atomicAdd(&cnt[i0], 1);
        list[i0 * BT + p0] = t;
        gwl [i0 * BT + p0] = w0;
        int p1 = atomicAdd(&cnt[i1], 1);
        list[i1 * BT + p1] = t;
        gwl [i1 * BT + p1] = w1;
    }
}

// ---------------------------------------------------------------------------
// Kernel 2: load-balance loss scalar
// ---------------------------------------------------------------------------
__global__ void moe_loss_kernel(const int* cnt, const float* probsum, float* out_loss) {
    if (blockIdx.x == 0 && threadIdx.x == 0) {
        float U = 0.0f, P = 0.0f;
#pragma unroll
        for (int e = 0; e < NEXP; ++e) { U += (float)cnt[e]; P += probsum[e]; }
        float l = 0.0f;
#pragma unroll
        for (int e = 0; e < NEXP; ++e)
            l += ((float)cnt[e] / U) * (probsum[e] / P);
        *out_loss = l * (float)NEXP;
    }
}

// ---------------------------------------------------------------------------
// Kernel 3: expert FFN.  grid = (token_tiles, E), block = 256 = 8 waves
// arranged 2 (M row-tiles) x 4 (N col-groups).  32 routed tokens per block.
//   GEMM1: h = x @ W1[e]^T   (K = D)   fused silu(h1)*h2 -> act (bf16, LDS)
//   GEMM2: y = act @ W2[e]^T (K = H)   12 accumulator tiles per wave
// PRE=true : weights pre-converted to bf16 (no conversion VALU in inner loop)
// PRE=false: fallback, convert fp32 weights in-register
// ---------------------------------------------------------------------------
template <bool PRE>
__global__ void __launch_bounds__(256)
moe_ffn_kernel(const float* __restrict__ x,
               const void*  __restrict__ W1p,
               const void*  __restrict__ W2p,
               const int*   __restrict__ cnt,
               const int*   __restrict__ list,
               const float* __restrict__ gwl,
               float* __restrict__ out) {
    const int e      = blockIdx.y;
    const int tstart = blockIdx.x * 32;
    const int total  = cnt[e];
    if (tstart >= total) return;
    const int tcount = min(32, total - tstart);

    const int tid  = threadIdx.x;
    const int wave = tid >> 5;
    const int lane = tid & 31;
    const int ln15 = lane & 15;
    const int hi   = (lane >> 4);        // 0: lanes 0-15, 1: lanes 16-31
    const int mi   = wave >> 2;          // 0..1  (row tile)
    const int ni   = wave & 3;           // 0..3  (col group)

    __shared__ unsigned short x_s[32 * DDIM];   // 48 KB bf16 token tile
    __shared__ unsigned short act_s[32 * 64];   //  4 KB bf16 activation chunk
    __shared__ int   tok_s[32];
    __shared__ float gw_s[32];

    if (tid < 32) {
        if (tid < tcount) {
            tok_s[tid] = list[e * BT + tstart + tid];
            gw_s[tid]  = gwl [e * BT + tstart + tid];
        } else {
            tok_s[tid] = 0;
            gw_s[tid]  = 0.0f;
        }
    }
    __syncthreads();

    // stage x tile (fp32 -> bf16), 32 rows x 768
    for (int i = tid; i < 32 * (DDIM / 4); i += 256) {
        int row = i / (DDIM / 4);
        int c4  = (i % (DDIM / 4)) * 4;
        float4 tv;
        if (row < tcount)
            tv = *reinterpret_cast<const float4*>(x + (size_t)tok_s[row] * DDIM + c4);
        else
            tv = make_float4(0.f, 0.f, 0.f, 0.f);
        unsigned short* dst = x_s + row * DDIM + c4;
        dst[0] = f2bf(tv.x); dst[1] = f2bf(tv.y);
        dst[2] = f2bf(tv.z); dst[3] = f2bf(tv.w);
    }
    __syncthreads();

    const unsigned short* W1b = nullptr;
    const unsigned short* W2b = nullptr;
    const float*          W1f = nullptr;
    const float*          W2f = nullptr;
    if constexpr (PRE) {
        W1b = (const unsigned short*)W1p + (size_t)e * (2 * HDIM) * DDIM;
        W2b = (const unsigned short*)W2p + (size_t)e * DDIM * HDIM;
    } else {
        W1f = (const float*)W1p + (size_t)e * (2 * HDIM) * DDIM;
        W2f = (const float*)W2p + (size_t)e * DDIM * HDIM;
    }

    // y accumulators: 12 tiles of 16 cols per wave (4 groups * 12 * 16 = 768)
    v8f accy[12];
#pragma unroll
    for (int j = 0; j < 12; ++j) accy[j] = (v8f){0.f,0.f,0.f,0.f,0.f,0.f,0.f,0.f};

    const unsigned short* a1base = x_s  + (mi * 16 + ln15) * DDIM;
    const unsigned short* a2base = act_s + (mi * 16 + ln15) * 64;

    for (int c = 0; c < HDIM / 64; ++c) {          // 24 chunks of 64 h-cols
        // ---- GEMM1: this wave's 16 h-columns (f) and paired gate cols (f+H)
        const int f = c * 64 + ni * 16;
        v8f acc1 = (v8f){0.f,0.f,0.f,0.f,0.f,0.f,0.f,0.f};
        v8f acc2 = (v8f){0.f,0.f,0.f,0.f,0.f,0.f,0.f,0.f};
        const size_t o1 = (size_t)(f + ln15) * DDIM        + hi * 16;
        const size_t o2 = (size_t)(HDIM + f + ln15) * DDIM + hi * 16;
#pragma unroll 4
        for (int kk = 0; kk < DDIM / 32; ++kk) {
            const int kb = kk * 32;
            v16bf a = load_a_lds(a1base, kb + hi * 16);
            v16bf b1, b2;
            if constexpr (PRE) {
                b1 = load_b_bf16(W1b + o1 + kb);
                b2 = load_b_bf16(W1b + o2 + kb);
            } else {
                b1 = load_b_f32(W1f + o1 + kb);
                b2 = load_b_f32(W1f + o2 + kb);
            }
            acc1 = __builtin_amdgcn_wmma_f32_16x16x32_bf16(false, a, false, b1,
                                                           (short)0, acc1, false, false);
            acc2 = __builtin_amdgcn_wmma_f32_16x16x32_bf16(false, a, false, b2,
                                                           (short)0, acc2, false, false);
        }

        // ---- silu(h1)*h2 -> bf16 chunk in LDS
        __syncthreads();   // previous chunk's act reads complete
        {
            unsigned short* arow = act_s + (mi * 16) * 64 + ni * 16 + ln15;
#pragma unroll
            for (int r = 0; r < 8; ++r) {
                float h1 = acc1[r];
                float h2 = acc2[r];
                float sv = h1 / (1.0f + __expf(-h1));
                arow[(r + hi * 8) * 64] = f2bf(sv * h2);
            }
        }
        __syncthreads();

        // ---- GEMM2: accumulate y over this K-chunk of 64
#pragma unroll
        for (int kk = 0; kk < 2; ++kk) {
            const int kb = kk * 32;
            v16bf a2 = load_a_lds(a2base, kb + hi * 16);
#pragma unroll
            for (int j = 0; j < 12; ++j) {
                const int dbase = (ni * 12 + j) * 16;
                const size_t ow = (size_t)(dbase + ln15) * HDIM + c * 64 + kb + hi * 16;
                v16bf b;
                if constexpr (PRE) b = load_b_bf16(W2b + ow);
                else               b = load_b_f32(W2f + ow);
                accy[j] = __builtin_amdgcn_wmma_f32_16x16x32_bf16(false, a2, false, b,
                                                                  (short)0, accy[j],
                                                                  false, false);
            }
        }
    }

    // ---- scale by gate weight and scatter-accumulate into output
#pragma unroll
    for (int j = 0; j < 12; ++j) {
        const int n = (ni * 12 + j) * 16 + ln15;
#pragma unroll
        for (int r = 0; r < 8; ++r) {
            const int m = mi * 16 + r + hi * 8;
            if (m < tcount) {
                atomicAdd(&out[(size_t)tok_s[m] * DDIM + n], accy[j][r] * gw_s[m]);
            }
        }
    }
}

// ---------------------------------------------------------------------------
extern "C" void kernel_launch(void* const* d_in, const int* in_sizes, int n_in,
                              void* d_out, int out_size, void* d_ws, size_t ws_size,
                              hipStream_t stream) {
    const float* x  = (const float*)d_in[0];
    const float* Wg = (const float*)d_in[1];
    const float* W1 = (const float*)d_in[2];
    const float* W2 = (const float*)d_in[3];
    const float* eb = (const float*)d_in[4];
    float* out = (float*)d_out;

    // workspace layout
    char*  ws      = (char*)d_ws;
    int*   cnt     = (int*)ws;                                  // 8 ints
    float* probsum = (float*)(ws + 64);                         // 8 floats
    int*   list    = (int*)(ws + 128);                          // E*BT ints
    float* gwl     = (float*)(ws + 128 + NEXP * BT * sizeof(int));
    const size_t bf_off = 131200;                               // 128-aligned
    unsigned short* W1b = (unsigned short*)(ws + bf_off);
    unsigned short* W2b = W1b + W1N;
    const size_t need = bf_off + (W1N + W2N) * sizeof(unsigned short);

    const int n_out_elems = BT * DDIM;

    moe_zero_kernel<<<1024, 256, 0, stream>>>(out, n_out_elems, cnt, probsum);
    moe_gate_kernel<<<BT / 8, 256, 0, stream>>>(x, Wg, eb, cnt, probsum, list, gwl);
    moe_loss_kernel<<<1, 64, 0, stream>>>(cnt, probsum, out + n_out_elems);

    if (ws_size >= need) {
        moe_conv_kernel<<<2048, 256, 0, stream>>>(W1, W1b, (int)(W1N / 4));
        moe_conv_kernel<<<2048, 256, 0, stream>>>(W2, W2b, (int)(W2N / 4));
        moe_ffn_kernel<true><<<dim3(BT / 32, NEXP), 256, 0, stream>>>(
            x, W1b, W2b, cnt, list, gwl, out);
    } else {
        moe_ffn_kernel<false><<<dim3(BT / 32, NEXP), 256, 0, stream>>>(
            x, W1, W2, cnt, list, gwl, out);
    }
}
